// SwinTransformer_68247030334221
// MI455X (gfx1250) — matrix operand order repaired and checked by reading, not compiled, not run on hardware
//
#include <hip/hip_runtime.h>
#include <math.h>

typedef __attribute__((ext_vector_type(16))) _Float16 v16h;
typedef __attribute__((ext_vector_type(8)))  _Float16 v8h;
typedef __attribute__((ext_vector_type(8)))  float    v8f;

union V16U { v16h v; v8h h[2]; };

#define WS    7
#define NTOK  49

// =====================================================================
// Pad-and-convert: f32 [rows,K] -> f16 [rowsPad,Kpad], zero padded.
// =====================================================================
__global__ void convert_pad_kernel(const float* __restrict__ src,
                                   _Float16* __restrict__ dst,
                                   int rows, int K, int Kpad, long total)
{
  long idx = (long)blockIdx.x * blockDim.x + threadIdx.x;
  if (idx >= total) return;
  int k = (int)(idx % Kpad);
  long r = idx / Kpad;
  float v = (r < rows && k < K) ? src[r * K + k] : 0.f;
  dst[idx] = (_Float16)v;
}

// =====================================================================
// GEMM via WMMA, f16 in / f32 accum:
//   out[M,N] = act(A[M,K] @ Wh[Npad,K]^T + bias) (+ resid)
// Requirements: M % 32 == 0, K % 32 == 0, Wh padded to Npad=ceil32(N) rows.
// Each wave computes a 32x32 tile (2x2 of 16x16) -> 4 WMMA per K-step.
// Inner loop is branchless: 8x b128 loads + 4x v_wmma (+ prefetch).
// =====================================================================
__global__ __launch_bounds__(128)
void gemm_wmma_kernel(const _Float16* __restrict__ A, const _Float16* __restrict__ Wh,
                      const float* __restrict__ bias, const float* __restrict__ resid,
                      void* __restrict__ outp, int M, int N, int K, int gelu, int out16)
{
  const int lane   = threadIdx.x & 31;
  const int wave   = threadIdx.x >> 5;
  const int tilesN = (N + 31) >> 5;
  const long tiles = (long)(M >> 5) * tilesN;
  long tile = (long)blockIdx.x * 4 + wave;
  if (tile >= tiles) return;                 // wave-uniform exit
  const int tm = (int)(tile / tilesN);
  const int tn = (int)(tile % tilesN);
  const int hi = lane >> 4;                  // half-wave selector
  const int ml = lane & 15;

  const _Float16* __restrict__ arow0 = A  + (size_t)(tm * 32 + ml)      * K;
  const _Float16* __restrict__ arow1 = A  + (size_t)(tm * 32 + 16 + ml) * K;
  const _Float16* __restrict__ brow0 = Wh + (size_t)(tn * 32 + ml)      * K;
  const _Float16* __restrict__ brow1 = Wh + (size_t)(tn * 32 + 16 + ml) * K;

  v8f c00 = {}, c01 = {}, c10 = {}, c11 = {};
  const int a_off = 8 * hi;                  // A: halves [k0+8hi..+7],[k0+16+8hi..+7]
  const int b_off = 8 * hi;                  // B: halves [k0+8hi..+7],[k0+16+8hi..+7]

  for (int k0 = 0; k0 < K; k0 += 32) {
    if (k0 + 64 <= K) {                      // prefetch next chunk (global_prefetch_b8)
      __builtin_prefetch(arow0 + k0 + 32 + a_off, 0, 1);
      __builtin_prefetch(brow0 + k0 + 32 + b_off, 0, 1);
    }
    V16U a0, a1, b0, b1;
    a0.h[0] = *(const v8h*)(arow0 + k0 + a_off);
    a0.h[1] = *(const v8h*)(arow0 + k0 + 16 + a_off);
    a1.h[0] = *(const v8h*)(arow1 + k0 + a_off);
    a1.h[1] = *(const v8h*)(arow1 + k0 + 16 + a_off);
    b0.h[0] = *(const v8h*)(brow0 + k0 + b_off);
    b0.h[1] = *(const v8h*)(brow0 + k0 + 16 + b_off);
    b1.h[0] = *(const v8h*)(brow1 + k0 + b_off);
    b1.h[1] = *(const v8h*)(brow1 + k0 + 16 + b_off);
    c00 = __builtin_amdgcn_wmma_f32_16x16x32_f16(false, a0.v, false, b0.v, (short)0, c00, false, false);
    c01 = __builtin_amdgcn_wmma_f32_16x16x32_f16(false, a0.v, false, b1.v, (short)0, c01, false, false);
    c10 = __builtin_amdgcn_wmma_f32_16x16x32_f16(false, a1.v, false, b0.v, (short)0, c10, false, false);
    c11 = __builtin_amdgcn_wmma_f32_16x16x32_f16(false, a1.v, false, b1.v, (short)0, c11, false, false);
  }

  auto emit = [&](const v8f& cc, int rbase, int cbase) {
    int col = cbase + ml;
    if (col < N) {
      float bv = bias ? bias[col] : 0.f;
#pragma unroll
      for (int j = 0; j < 8; ++j) {
        int row = rbase + j + 8 * hi;        // C/D layout: VGPR j -> M = j + 8*hi
        float v = cc[j] + bv;
        if (gelu) v = 0.5f * v * (1.f + erff(v * 0.70710678118654752f));
        if (resid) v += resid[(size_t)row * N + col];
        if (out16) ((_Float16*)outp)[(size_t)row * N + col] = (_Float16)v;
        else       ((float*)outp)[(size_t)row * N + col] = v;
      }
    }
  };
  emit(c00, tm * 32,      tn * 32);
  emit(c01, tm * 32,      tn * 32 + 16);
  emit(c10, tm * 32 + 16, tn * 32);
  emit(c11, tm * 32 + 16, tn * 32 + 16);
}

// =====================================================================
// LayerNorm helpers (blockDim == 128); templated output type
// =====================================================================
template<typename DstT>
__device__ __forceinline__ void ln_row(const float* __restrict__ src,
                                       DstT* __restrict__ dst,
                                       const float* __restrict__ g,
                                       const float* __restrict__ b, int C)
{
  __shared__ float r1[128], r2[128];
  int tid = threadIdx.x;
  float s = 0.f, s2 = 0.f;
  for (int c = tid; c < C; c += 128) { float x = src[c]; s += x; s2 += x * x; }
  r1[tid] = s; r2[tid] = s2; __syncthreads();
  for (int off = 64; off > 0; off >>= 1) {
    if (tid < off) { r1[tid] += r1[tid + off]; r2[tid] += r2[tid + off]; }
    __syncthreads();
  }
  float mean = r1[0] / C;
  float var  = r2[0] / C - mean * mean;
  float rstd = rsqrtf(var + 1e-5f);
  for (int c = tid; c < C; c += 128)
    dst[c] = (DstT)((src[c] - mean) * rstd * g[c] + b[c]);
}

template<typename DstT>
__global__ __launch_bounds__(128)
void ln_kernel(const float* __restrict__ in, const float* __restrict__ g,
               const float* __restrict__ b, DstT* __restrict__ out, int C)
{
  size_t row = blockIdx.x;
  ln_row<DstT>(in + row * C, out + row * C, g, b, C);
}

// LN + cyclic shift + window partition -> f16 rows [(b*nWin+wIdx)*49+p, C]
__global__ __launch_bounds__(128)
void ln_window_kernel(const float* __restrict__ z, const float* __restrict__ g,
                      const float* __restrict__ b, _Float16* __restrict__ out,
                      int H, int Wd, int C, int shift)
{
  int o = blockIdx.x;
  int p = o % NTOK, bwin = o / NTOK;
  int nWw = Wd / WS, nWin = (H / WS) * nWw;
  int wIdx = bwin % nWin, bb = bwin / nWin;
  int hh = (wIdx / nWw) * WS + p / WS;
  int ww = (wIdx % nWw) * WS + p % WS;
  int sh = (hh + shift) % H;                 // roll(z, -shift)
  int sw = (ww + shift) % Wd;
  const float* src = z + (((size_t)bb * H + sh) * Wd + sw) * C;
  ln_row<_Float16>(src, out + (size_t)o * C, g, b, C);
}

// Merge: gather 2x2 neighborhood concat (4C) + LN -> f16
__global__ __launch_bounds__(128)
void merge_ln_kernel(const float* __restrict__ z, const float* __restrict__ g,
                     const float* __restrict__ b, _Float16* __restrict__ out,
                     int H, int Wd, int C)
{
  int o = blockIdx.x;                        // rows over B*(H/2)*(W/2)
  int Wo = Wd >> 1, Ho = H >> 1;
  int j = o % Wo; int t = o / Wo; int i2 = t % Ho; int bb = t / Ho;
  int C4 = 4 * C;
  auto val = [&](int c) -> float {
    int q = c / C, cc = c % C;
    int h = 2 * i2 + (q & 1);                // order: (0,0),(1,0),(0,1),(1,1)
    int w = 2 * j + (q >> 1);
    return z[(((size_t)bb * H + h) * Wd + w) * C + cc];
  };
  __shared__ float r1[128], r2[128];
  int tid = threadIdx.x;
  float s = 0.f, s2 = 0.f;
  for (int c = tid; c < C4; c += 128) { float x = val(c); s += x; s2 += x * x; }
  r1[tid] = s; r2[tid] = s2; __syncthreads();
  for (int off = 64; off > 0; off >>= 1) {
    if (tid < off) { r1[tid] += r1[tid + off]; r2[tid] += r2[tid + off]; }
    __syncthreads();
  }
  float mean = r1[0] / C4;
  float var  = r2[0] / C4 - mean * mean;
  float rstd = rsqrtf(var + 1e-5f);
  _Float16* drow = out + (size_t)o * C4;
  for (int c = tid; c < C4; c += 128)
    drow[c] = (_Float16)((val(c) - mean) * rstd * g[c] + b[c]);
}

// =====================================================================
// Window attention via WMMA: one block (128 thr / 4 waves) per (window, head).
// 49 tokens padded to 64; hd=32.
//   S = Q K^T : 16 WMMA tiles (16x16x32), operands from LDS (ds_load_b128)
//   softmax(+rpb bias, +shift mask) in f32 LDS
//   O = P V   : 8 tiles x 2 K-steps = 16 WMMAs, V pre-transposed in LDS
// =====================================================================
__device__ __forceinline__ int region7(int h, int Hs, int shift) {
  return (h < Hs - WS) ? 0 : ((h < Hs - shift) ? 1 : 2);
}

__global__ __launch_bounds__(128)
void attn_kernel(const _Float16* __restrict__ qkv, const float* __restrict__ rpb,
                 _Float16* __restrict__ out, int C, int nh, int H, int Wd, int shift)
{
  __shared__ __align__(16) _Float16 Qh[64 * 32];   // Q padded, pre-scaled
  __shared__ __align__(16) _Float16 Kh[64 * 32];   // K padded
  __shared__ __align__(16) _Float16 Vt[32 * 64];   // V transposed [d][j]
  __shared__ __align__(16) _Float16 Ph[64 * 64];   // softmax probs (f16)
  __shared__ __align__(16) float    Sf[64 * 64];   // raw scores (f32)
  __shared__ float bias_s[(2 * WS - 1) * (2 * WS - 1)];

  int gid = blockIdx.x;
  int head = gid % nh, bwin = gid / nh;
  int nWw = Wd / WS, nWin = (H / WS) * nWw;
  int wIdx = bwin % nWin;
  int wi_h = wIdx / nWw, wi_w = wIdx % nWw;
  int tid = threadIdx.x;
  int lane = tid & 31, wave = tid >> 5;
  int hi = lane >> 4, ml = lane & 15;
  int C3 = 3 * C;
  size_t base = (size_t)bwin * NTOK;

  // ---- stage q,k,v (f16) into LDS with zero padding; v transposed ----
  for (int t = tid; t < 64 * 32; t += 128) {
    int i = t >> 5, d = t & 31;
    _Float16 qv = (_Float16)0.f, kv = (_Float16)0.f, vv = (_Float16)0.f;
    if (i < NTOK) {
      const _Float16* row = qkv + (base + i) * C3;
      qv = (_Float16)((float)row[(0 * nh + head) * 32 + d] * 0.17677669529663687f);
      kv = row[(1 * nh + head) * 32 + d];
      vv = row[(2 * nh + head) * 32 + d];
    }
    Qh[i * 32 + d] = qv;
    Kh[i * 32 + d] = kv;
    Vt[d * 64 + i] = vv;
  }
  for (int t = tid; t < (2 * WS - 1) * (2 * WS - 1); t += 128)
    bias_s[t] = rpb[t * nh + head];
  __syncthreads();

  // ---- S = Q K^T : wave handles tiles wave, wave+4, wave+8, wave+12 ----
  for (int tt = wave; tt < 16; tt += 4) {
    int tm = tt >> 2, tn = tt & 3;
    V16U a, b;
    const _Float16* qrow = Qh + (tm * 16 + ml) * 32;
    const _Float16* krow = Kh + (tn * 16 + ml) * 32;
    a.h[0] = *(const v8h*)(qrow + 8 * hi);
    a.h[1] = *(const v8h*)(qrow + 16 + 8 * hi);
    b.h[0] = *(const v8h*)(krow + 8 * hi);
    b.h[1] = *(const v8h*)(krow + 16 + 8 * hi);
    v8f c = {};
    c = __builtin_amdgcn_wmma_f32_16x16x32_f16(false, a.v, false, b.v, (short)0, c, false, false);
#pragma unroll
    for (int j = 0; j < 8; ++j)
      Sf[(tm * 16 + j + 8 * hi) * 64 + tn * 16 + ml] = c[j];
  }
  __syncthreads();

  // ---- bias + mask + softmax (rows handled by threads 0..63) ----
  if (tid < 64) {
    int i = tid;
    if (i < NTOK) {
      int yi = i / WS, xi = i % WS;
      int ri = 0;
      if (shift > 0)
        ri = region7(wi_h * WS + yi, H, shift) * 3 + region7(wi_w * WS + xi, Wd, shift);
      float m = -1e30f;
      for (int j = 0; j < NTOK; ++j) {
        int yj = j / WS, xj = j % WS;
        int rel = (yi - yj + WS - 1) * (2 * WS - 1) + (xi - xj + WS - 1);
        float s = Sf[i * 64 + j] + bias_s[rel];
        if (shift > 0) {
          int rj = region7(wi_h * WS + yj, H, shift) * 3 + region7(wi_w * WS + xj, Wd, shift);
          if (ri != rj) s -= 100.f;
        }
        Sf[i * 64 + j] = s;
        m = fmaxf(m, s);
      }
      float sum = 0.f;
      for (int j = 0; j < NTOK; ++j) { float e = __expf(Sf[i * 64 + j] - m); Sf[i * 64 + j] = e; sum += e; }
      float inv = 1.f / sum;
      for (int j = 0; j < NTOK; ++j) Ph[i * 64 + j] = (_Float16)(Sf[i * 64 + j] * inv);
      for (int j = NTOK; j < 64; ++j) Ph[i * 64 + j] = (_Float16)0.f;
    } else {
      for (int j = 0; j < 64; ++j) Ph[i * 64 + j] = (_Float16)0.f;
    }
  }
  __syncthreads();

  // ---- O = P V : tile rows tm=wave (0..3), tn in {0,1}; K=64 ----
  {
    int tm = wave;
    for (int tn = 0; tn < 2; ++tn) {
      v8f c = {};
      for (int k0 = 0; k0 < 64; k0 += 32) {
        V16U a, b;
        const _Float16* prow = Ph + (tm * 16 + ml) * 64 + k0;
        const _Float16* vrow = Vt + (tn * 16 + ml) * 64 + k0;
        a.h[0] = *(const v8h*)(prow + 8 * hi);
        a.h[1] = *(const v8h*)(prow + 16 + 8 * hi);
        b.h[0] = *(const v8h*)(vrow + 8 * hi);
        b.h[1] = *(const v8h*)(vrow + 16 + 8 * hi);
        c = __builtin_amdgcn_wmma_f32_16x16x32_f16(false, a.v, false, b.v, (short)0, c, false, false);
      }
#pragma unroll
      for (int j = 0; j < 8; ++j) {
        int row = tm * 16 + j + 8 * hi;
        if (row < NTOK)
          out[(base + row) * C + head * 32 + tn * 16 + ml] = (_Float16)c[j];
      }
    }
  }
}

// window reverse + un-shift + residual add into z (win = proj output, f32)
__global__ void wrev_add_kernel(const float* __restrict__ win, float* __restrict__ z,
                                int H, int Wd, int C, int shift, long total)
{
  long idx = (long)blockIdx.x * blockDim.x + threadIdx.x;
  if (idx >= total) return;
  int c = (int)(idx % C); long r = idx / C;
  int w = (int)(r % Wd); r /= Wd; int h = (int)(r % H); int bb = (int)(r / H);
  int hp = (h - shift + H) % H;              // coords in rolled frame
  int wp = (w - shift + Wd) % Wd;
  int nWw = Wd / WS, nWin = (H / WS) * nWw;
  int wIdx = (hp / WS) * nWw + (wp / WS);
  int p = (hp % WS) * WS + (wp % WS);
  z[idx] += win[(((size_t)bb * nWin + wIdx) * NTOK + p) * C + c];
}

// patch embed gather: x[B,3,224,224] -> xp f16 [B*56*56, 64] (K padded 48->64)
__global__ void patch_gather_kernel(const float* __restrict__ x,
                                    _Float16* __restrict__ out, long total)
{
  long idx = (long)blockIdx.x * blockDim.x + threadIdx.x;
  if (idx >= total) return;
  int c = (int)(idx % 64); long r = idx / 64;
  int w = (int)(r % 56); r /= 56; int h = (int)(r % 56); int bb = (int)(r / 56);
  float v = 0.f;
  if (c < 48) {
    int ch = c / 16, py = (c % 16) / 4, px = c % 4;
    v = x[(((size_t)bb * 3 + ch) * 224 + h * 4 + py) * 224 + w * 4 + px];
  }
  out[idx] = (_Float16)v;
}

// mean over tokens: f16 in -> f16 out
__global__ void pool_kernel(const _Float16* __restrict__ in, _Float16* __restrict__ out,
                            int T, int C, long total)
{
  long idx = (long)blockIdx.x * blockDim.x + threadIdx.x;
  if (idx >= total) return;
  int c = (int)(idx % C); int bb = (int)(idx / C);
  float s = 0.f;
  for (int t = 0; t < T; ++t) s += (float)in[((size_t)bb * T + t) * C + c];
  out[idx] = (_Float16)(s / (float)T);
}

// =====================================================================
// Host side
// =====================================================================
static const int DEPTHS_[4] = {2, 2, 6, 2};
static const int HEADS_[4]  = {3, 6, 12, 24};

struct BlockP {
  const float *n1g, *n1b, *qkv_w, *qkv_b, *rpb, *proj_w, *proj_b,
              *n2g, *n2b, *fc1_w, *fc1_b, *fc2_w, *fc2_b;
};
struct MergeP { const float *ng, *nb, *w; };
struct NetP {
  const float *pe_w, *pe_b, *pe_ng, *pe_nb;
  BlockP blk[12];
  MergeP mg[3];
  const float *ng, *nb, *head_w, *head_b, *x;
};

static void unpack_params(void* const* d_in, const int* in_sizes, NetP& P)
{
  auto F = [&](int& i) { return (const float*)d_in[i++]; };
  int i = 0;
  if (in_sizes[0] == 1000) {
    // jax tree_leaves: dicts in sorted-key order; top: params then x
    P.head_b = F(i); P.head_w = F(i);
    int bi = 0;
    for (int li = 0; li < 4; ++li) {
      for (int d = 0; d < DEPTHS_[li]; ++d) {
        BlockP& B = P.blk[bi++];
        B.fc1_b = F(i); B.fc1_w = F(i); B.fc2_b = F(i); B.fc2_w = F(i);
        B.n1b = F(i); B.n1g = F(i); B.n2b = F(i); B.n2g = F(i);
        B.proj_b = F(i); B.proj_w = F(i); B.qkv_b = F(i); B.qkv_w = F(i);
        B.rpb = F(i);
      }
      if (li < 3) { P.mg[li].nb = F(i); P.mg[li].ng = F(i); P.mg[li].w = F(i); }
    }
    P.nb = F(i); P.ng = F(i);
    P.pe_b = F(i); P.pe_nb = F(i); P.pe_ng = F(i); P.pe_w = F(i);
    P.x = F(i);
  } else {
    // insertion order; x may come first (top-level dict order) or last
    bool x_first = (in_sizes[0] != 4608);
    if (x_first) P.x = F(i);
    P.pe_w = F(i); P.pe_b = F(i); P.pe_ng = F(i); P.pe_nb = F(i);
    int bi = 0;
    for (int li = 0; li < 4; ++li) {
      for (int d = 0; d < DEPTHS_[li]; ++d) {
        BlockP& B = P.blk[bi++];
        B.n1g = F(i); B.n1b = F(i); B.qkv_w = F(i); B.qkv_b = F(i); B.rpb = F(i);
        B.proj_w = F(i); B.proj_b = F(i); B.n2g = F(i); B.n2b = F(i);
        B.fc1_w = F(i); B.fc1_b = F(i); B.fc2_w = F(i); B.fc2_b = F(i);
      }
      if (li < 3) { P.mg[li].ng = F(i); P.mg[li].nb = F(i); P.mg[li].w = F(i); }
    }
    P.ng = F(i); P.nb = F(i); P.head_w = F(i); P.head_b = F(i);
    if (!x_first) P.x = F(i);
  }
}

static inline void run_gemm(const _Float16* A, const _Float16* Wh, const float* bias,
                            const float* resid, void* out, int M, int N, int K,
                            int gelu, int out16, hipStream_t s)
{
  long tiles = (long)(M / 32) * ((N + 31) / 32);
  int blocks = (int)((tiles + 3) / 4);
  gemm_wmma_kernel<<<blocks, 128, 0, s>>>(A, Wh, bias, resid, out, M, N, K, gelu, out16);
}

extern "C" void kernel_launch(void* const* d_in, const int* in_sizes, int n_in,
                              void* d_out, int out_size, void* d_ws, size_t ws_size,
                              hipStream_t stream)
{
  (void)n_in; (void)out_size; (void)ws_size;
  NetP P;
  unpack_params(d_in, in_sizes, P);

  const int B = 64;
  // ---- workspace layout (bytes) ----
  // z   : f32 [200704*96]            = 77,070,336
  // tA  : f16 [200704*96]            = 38,535,168
  // big : f16 [200704*384] (or f32 view [200704*96]) = 154,140,672
  // w16 : converted weights (~57 MB)
  char* ws = (char*)d_ws;
  float*    z   = (float*)ws;
  _Float16* tA  = (_Float16*)(ws + 77070336);
  char*     bigRaw = ws + 115605504;
  _Float16* big16  = (_Float16*)bigRaw;
  float*    big32  = (float*)bigRaw;
  _Float16* wcur   = (_Float16*)(ws + 269746176);

  // ---- convert all weights to f16 (padded) ----
  auto conv = [&](const float* src, int rows, int K, int Kpad, int rowsPad) -> _Float16* {
    _Float16* dst = wcur;
    long total = (long)rowsPad * Kpad;
    convert_pad_kernel<<<(int)((total + 255) / 256), 256, 0, stream>>>(src, dst, rows, K, Kpad, total);
    wcur += total;
    return dst;
  };
  _Float16* pe_w16 = conv(P.pe_w, 96, 48, 64, 96);
  _Float16 *qkv16[12], *proj16[12], *fc116[12], *fc216[12], *mg16[3];
  {
    int bi = 0;
    for (int li = 0; li < 4; ++li) {
      int C = 96 << li;
      for (int d = 0; d < DEPTHS_[li]; ++d) {
        qkv16[bi]  = conv(P.blk[bi].qkv_w,  3 * C, C,     C,     3 * C);
        proj16[bi] = conv(P.blk[bi].proj_w, C,     C,     C,     C);
        fc116[bi]  = conv(P.blk[bi].fc1_w,  4 * C, C,     C,     4 * C);
        fc216[bi]  = conv(P.blk[bi].fc2_w,  C,     4 * C, 4 * C, C);
        ++bi;
      }
      if (li < 3) mg16[li] = conv(P.mg[li].w, 2 * C, 4 * C, 4 * C, 2 * C);
    }
  }
  _Float16* head16 = conv(P.head_w, 1000, 768, 768, 1024);

  // ---- patch embed: gather(f16, K pad 64) -> GEMM -> LN(f32 -> z) ----
  {
    long total = (long)200704 * 64;
    patch_gather_kernel<<<(int)((total + 255) / 256), 256, 0, stream>>>(P.x, tA, total);
    run_gemm(tA, pe_w16, P.pe_b, nullptr, big32, 200704, 96, 64, 0, 0, stream);
    ln_kernel<float><<<200704, 128, 0, stream>>>(big32, P.pe_ng, P.pe_nb, z, 96);
  }

  int H = 56, Wd = 56, C = 96, blkIdx = 0;
  for (int li = 0; li < 4; ++li) {
    const int nh = HEADS_[li];
    const int M = B * H * Wd;
    const int nWin = (H / WS) * (Wd / WS);
    for (int d = 0; d < DEPTHS_[li]; ++d) {
      const BlockP& bp = P.blk[blkIdx];
      const int shift = ((d & 1) && li < 3) ? (WS / 2) : 0;
      // LN1 + shift + window partition (f16)
      ln_window_kernel<<<M, 128, 0, stream>>>(z, bp.n1g, bp.n1b, tA, H, Wd, C, shift);
      // QKV (f16 out)
      run_gemm(tA, qkv16[blkIdx], bp.qkv_b, nullptr, big16, M, 3 * C, C, 0, 1, stream);
      // attention (WMMA) -> tA (f16)
      attn_kernel<<<B * nWin * nh, 128, 0, stream>>>(big16, bp.rpb, tA, C, nh, H, Wd, shift);
      // proj -> big32 (f32)
      run_gemm(tA, proj16[blkIdx], bp.proj_b, nullptr, big32, M, C, C, 0, 0, stream);
      // window reverse + unshift + residual add into z
      {
        long total = (long)M * C;
        wrev_add_kernel<<<(int)((total + 255) / 256), 256, 0, stream>>>(big32, z, H, Wd, C, shift, total);
      }
      // LN2 (f16) -> fc1(GELU, f16) -> fc2(+resid, f32 in-place on z)
      ln_kernel<_Float16><<<M, 128, 0, stream>>>(z, bp.n2g, bp.n2b, tA, C);
      run_gemm(tA, fc116[blkIdx], bp.fc1_b, nullptr, big16, M, 4 * C, C, 1, 1, stream);
      run_gemm(big16, fc216[blkIdx], bp.fc2_b, z, z, M, C, 4 * C, 0, 0, stream);
      ++blkIdx;
    }
    if (li < 3) {
      merge_ln_kernel<<<M / 4, 128, 0, stream>>>(z, P.mg[li].ng, P.mg[li].nb, tA, H, Wd, C);
      run_gemm(tA, mg16[li], nullptr, nullptr, z, M / 4, 2 * C, 4 * C, 0, 0, stream);
      H >>= 1; Wd >>= 1; C <<= 1;
    }
  }

  // ---- final LN (f16), token mean pool (f16), classifier head (f32 out) ----
  ln_kernel<_Float16><<<B * NTOK, 128, 0, stream>>>(z, P.ng, P.nb, tA, 768);
  _Float16* pooled = tA + 4000000;           // past 3136*768 rows, 16B-aligned
  {
    long total = (long)B * 768;
    pool_kernel<<<(int)((total + 255) / 256), 256, 0, stream>>>(tA, pooled, NTOK, 768, total);
  }
  run_gemm(pooled, head16, P.head_b, nullptr, (float*)d_out, B, 1000, 768, 0, 0, stream);
}
